// LocalTransformerEncoderLayer_15135464751246
// MI455X (gfx1250) — matrix operand order repaired and checked by reading, not compiled, not run on hardware
//
#include <hip/hip_runtime.h>

#define DEV __device__ __forceinline__

typedef __attribute__((ext_vector_type(16))) __bf16 bf16x16;
typedef __attribute__((ext_vector_type(8)))  float  floatx8;

union Frag { unsigned int u32[8]; bf16x16 v; };

// ---- helpers ---------------------------------------------------------------

DEV unsigned short f2bf(float x) {           // f32 -> bf16, round-to-nearest-even
  unsigned int u = __float_as_uint(x);
  unsigned int r = ((u >> 16) & 1u) + 0x7FFFu;
  return (unsigned short)((u + r) >> 16);
}

// Generic pointer to a __shared__ object: low 32 bits are the LDS byte offset
// (LDS aperture truncation, ISA 10.2).
DEV unsigned lds_off(const void* p) { return (unsigned)(unsigned long long)p; }

// gfx1250 async copy: 16 bytes per lane, global -> LDS, tracked by ASYNCcnt.
DEV void async_b128(unsigned dst_lds, const void* src_global) {
  asm volatile("global_load_async_to_lds_b128 %0, %1, off"
               :: "v"(dst_lds), "v"(src_global) : "memory");
}
DEV void wait_async0() { asm volatile("s_wait_asynccnt 0" ::: "memory"); }

// A-matrix 16x32 bf16 fragment (ISA 7.12.2): lane m=lane&15, h=lane>>4.
// VGPR r<4: K = 2r+8h,2r+1+8h ; r>=4: K = 16+2(r-4)+8h, +1
DEV void load_frag_a(Frag& f, const unsigned short* lds, int row0, int stride, int lane) {
  int m = lane & 15, h = (lane >> 4) & 1;
  const unsigned short* base = lds + (size_t)(row0 + m) * stride;
#pragma unroll
  for (int r = 0; r < 8; ++r) {
    int kb = (r < 4) ? (2 * r + 8 * h) : (2 * r + 8 + 8 * h);
    f.u32[r] = *(const unsigned int*)(base + kb);
  }
}

// B-matrix 32x16 bf16 fragment, LDS holds B transposed: bt[n][kk].
// lane n=lane&15, h=lane>>4; VGPR r<4: K = 2r+8h,2r+1+8h ; r>=4: K = 16+2(r-4)+8h, +1
DEV void load_frag_b(Frag& f, const unsigned short* lds, int col0, int stride, int lane) {
  int n = lane & 15, h = (lane >> 4) & 1;
  const unsigned short* base = lds + (size_t)(col0 + n) * stride;
#pragma unroll
  for (int r = 0; r < 8; ++r)
    f.u32[r] = *(const unsigned int*)(base + ((r < 4) ? (2 * r + 8 * h) : (2 * r + 8 + 8 * h)));
}

DEV floatx8 wmma_bf16(const Frag& a, const Frag& b, floatx8 c) {
  return __builtin_amdgcn_wmma_f32_16x16x32_bf16(false, a.v, false, b.v,
                                                 (short)0, c, false, false);
}

// ---- generic GEMM: out = epi(A @ W + bias) ---------------------------------
// A: [M,K] (f32 or bf16), W: [K,N] f32 (converted to bf16 at stage),
// 128x128 tile per block, BK=32, 8 waves (wave grid 4x2, each 32x64 region).
// Register-staged double buffer: next K-chunk's global loads are issued right
// after the barrier so they overlap the WMMA work of the current chunk.
// EPI: 0 = bf16 out * oscale ; 1 = relu -> bf16 ; 2 = f32 out
template <int EPI, bool ABF16>
__global__ __launch_bounds__(256) void gemm_kernel(
    const void* __restrict__ Ap, const float* __restrict__ Wp,
    const float* __restrict__ biasp, void* __restrict__ outp,
    int K, int N, float oscale)
{
  __shared__ alignas(16) unsigned short sA[128 * 32];
  __shared__ alignas(16) unsigned short sB[128 * 32];   // transposed: [n][kk]
  const int tid = threadIdx.x;
  const int lane = tid & 31, wave = tid >> 5;
  const int bm = blockIdx.x * 128, bn = blockIdx.y * 128;
  const int wr = wave & 3, wc = wave >> 2;
  const int m0 = wr * 32, n0 = wc * 64;

  const floatx8 z = {0.f, 0.f, 0.f, 0.f, 0.f, 0.f, 0.f, 0.f};
  floatx8 acc[2][4];
#pragma unroll
  for (int a = 0; a < 2; ++a)
#pragma unroll
    for (int b = 0; b < 4; ++b) acc[a][b] = z;

  const int arow = tid >> 1, akk = (tid & 1) * 16;   // A stage: 16 elems/thread
  const int bkk = tid >> 3, bnn = (tid & 7) * 16;    // B stage: 16 elems/thread

  const unsigned short* A16 = (const unsigned short*)Ap;
  const float* A32 = (const float*)Ap;
  float aRf[16]; unsigned short aRh[16]; float bRf[16];

  auto fetch = [&](int k0) {
    if (ABF16) {
      const unsigned short* g = A16 + (size_t)(bm + arow) * K + k0 + akk;
#pragma unroll
      for (int i = 0; i < 16; ++i) aRh[i] = g[i];
    } else {
      const float* g = A32 + (size_t)(bm + arow) * K + k0 + akk;
#pragma unroll
      for (int i = 0; i < 16; ++i) aRf[i] = g[i];
    }
    const float* gw = Wp + (size_t)(k0 + bkk) * N + bn + bnn;
#pragma unroll
    for (int i = 0; i < 16; ++i) bRf[i] = gw[i];
  };
  auto commit = [&]() {
    unsigned short* d = sA + arow * 32 + akk;
    if (ABF16) {
#pragma unroll
      for (int i = 0; i < 16; ++i) d[i] = aRh[i];
    } else {
#pragma unroll
      for (int i = 0; i < 16; ++i) d[i] = f2bf(aRf[i]);
    }
#pragma unroll
    for (int i = 0; i < 16; ++i) sB[(bnn + i) * 32 + bkk] = f2bf(bRf[i]);
  };

  fetch(0);
  for (int k0 = 0; k0 < K; k0 += 32) {
    commit();
    __syncthreads();
    if (k0 + 32 < K) fetch(k0 + 32);        // in flight during the WMMAs below
    Frag af[2];
    load_frag_a(af[0], sA, m0, 32, lane);
    load_frag_a(af[1], sA, m0 + 16, 32, lane);
#pragma unroll
    for (int tc = 0; tc < 4; ++tc) {
      Frag bf_;
      load_frag_b(bf_, sB, n0 + tc * 16, 32, lane);
      acc[0][tc] = wmma_bf16(af[0], bf_, acc[0][tc]);
      acc[1][tc] = wmma_bf16(af[1], bf_, acc[1][tc]);
    }
    __syncthreads();
  }

  const int h = (lane >> 4) & 1, nn = lane & 15;
#pragma unroll
  for (int tr = 0; tr < 2; ++tr)
#pragma unroll
    for (int tc = 0; tc < 4; ++tc) {
      int col = bn + n0 + tc * 16 + nn;
      float bv = biasp[col];
#pragma unroll
      for (int i = 0; i < 8; ++i) {
        size_t idx = (size_t)(bm + m0 + tr * 16 + h * 8 + i) * N + col;
        float v = acc[tr][tc][i] + bv;
        if (EPI == 0)      ((unsigned short*)outp)[idx] = f2bf(v * oscale);
        else if (EPI == 1) ((unsigned short*)outp)[idx] = f2bf(fmaxf(v, 0.0f));
        else               ((float*)outp)[idx] = v;
      }
    }
}

// ---- local attention: one block per (window, batch) ------------------------
struct AttnP1 { unsigned short qa[64 * 32]; unsigned short kt[192 * 32]; };
union AttnU { AttnP1 p1; unsigned short vt[64 * 192]; };

__global__ __launch_bounds__(256) void attn_kernel(
    const unsigned short* __restrict__ qb, const unsigned short* __restrict__ kb,
    const unsigned short* __restrict__ vb, const float* __restrict__ src,
    const unsigned char* __restrict__ maskp, float* __restrict__ outres)
{
  constexpr int Dm = 512, Wn = 64, Nn = 8192;
  const int w = blockIdx.x, b = blockIdx.y;
  const int tid = threadIdx.x, lane = tid & 31, wave = tid >> 5;
  const int h = (lane >> 4) & 1, nn = lane & 15;
  const int wr = wave & 3, wc = wave >> 2;     // phase 1/2 wave grid: 4 x 2
  const size_t bOff = (size_t)b * Nn;

  __shared__ alignas(16) AttnU u;                       // 24 KB (union)
  __shared__ alignas(16) unsigned short pLds[64 * 192]; // 24 KB probs (bf16)
  __shared__ float rowred[64 * 2];                      // per-row partials

  const floatx8 z = {0.f, 0.f, 0.f, 0.f, 0.f, 0.f, 0.f, 0.f};
  floatx8 acc[6];
#pragma unroll
  for (int t = 0; t < 6; ++t) acc[t] = z;

  // phase 1: sim(64x192) = Qw @ Kw^T, K streamed over d in chunks of 32.
  // q/k staging is a raw bf16 copy -> gfx1250 async global->LDS b128 path.
  const int qrow = tid >> 2, qkk = (tid & 3) * 8;      // 16 bytes per thread
  const unsigned qaB = lds_off(u.p1.qa);
  const unsigned ktB = lds_off(u.p1.kt);
  for (int d0 = 0; d0 < Dm; d0 += 32) {
    async_b128(qaB + (unsigned)(qrow * 32 + qkk) * 2,
               qb + (bOff + (size_t)w * Wn + qrow) * Dm + d0 + qkk);
#pragma unroll
    for (int s = 0; s < 3; ++s) {
      int task = s * 256 + tid;
      int j = task >> 2, kk0 = (task & 3) * 8;         // 16 bytes per task
      int gt = w * Wn - Wn + j;
      if (gt >= 0 && gt < Nn) {
        async_b128(ktB + (unsigned)(j * 32 + kk0) * 2,
                   kb + (bOff + gt) * Dm + d0 + kk0);
      } else {
        *(uint4*)&u.p1.kt[j * 32 + kk0] = make_uint4(0u, 0u, 0u, 0u);
      }
    }
    wait_async0();
    __syncthreads();
    Frag af;
    load_frag_a(af, u.p1.qa, wr * 16, 32, lane);
#pragma unroll
    for (int t = 0; t < 6; ++t) {
      Frag bf_;
      load_frag_b(bf_, u.p1.kt, (wc * 6 + t) * 16, 32, lane);
      acc[t] = wmma_bf16(af, bf_, acc[t]);
    }
    __syncthreads();
  }

  // phase 2: masked softmax over 192 keys (sim stays in registers)
#pragma unroll
  for (int t = 0; t < 6; ++t) {
    int j = (wc * 6 + t) * 16 + nn;
    int gt = w * Wn - Wn + j;
    bool ok = (gt >= 0) && (gt < Nn) && (maskp[bOff + gt] != 0);
    if (!ok) {
#pragma unroll
      for (int i = 0; i < 8; ++i) acc[t][i] = -1e10f;
    }
  }
  float rmax[8];
#pragma unroll
  for (int i = 0; i < 8; ++i) {
    float m = acc[0][i];
#pragma unroll
    for (int t = 1; t < 6; ++t) m = fmaxf(m, acc[t][i]);
#pragma unroll
    for (int off = 8; off >= 1; off >>= 1) m = fmaxf(m, __shfl_xor(m, off));
    rmax[i] = m;
  }
  if (nn == 0) {
#pragma unroll
    for (int i = 0; i < 8; ++i) rowred[(wr * 16 + h * 8 + i) * 2 + wc] = rmax[i];
  }
  __syncthreads();
  float mx[8];
#pragma unroll
  for (int i = 0; i < 8; ++i) {
    int r = wr * 16 + h * 8 + i;
    mx[i] = fmaxf(rowred[r * 2 + 0], rowred[r * 2 + 1]);
  }
  __syncthreads();
  float rsum[8];
#pragma unroll
  for (int i = 0; i < 8; ++i) {
    float s = 0.f;
#pragma unroll
    for (int t = 0; t < 6; ++t) s += __expf(acc[t][i] - mx[i]);
#pragma unroll
    for (int off = 8; off >= 1; off >>= 1) s += __shfl_xor(s, off);
    rsum[i] = s;
  }
  if (nn == 0) {
#pragma unroll
    for (int i = 0; i < 8; ++i) rowred[(wr * 16 + h * 8 + i) * 2 + wc] = rsum[i];
  }
  __syncthreads();
#pragma unroll
  for (int i = 0; i < 8; ++i) {
    int r = wr * 16 + h * 8 + i;
    float rinv = 1.0f / (rowred[r * 2 + 0] + rowred[r * 2 + 1]);
#pragma unroll
    for (int t = 0; t < 6; ++t) {
      int j = (wc * 6 + t) * 16 + nn;
      pLds[r * 192 + j] = f2bf(__expf(acc[t][i] - mx[i]) * rinv);
    }
  }
  __syncthreads();

  // phase 3: out(64x512) = P(64x192) @ V(192x512), 64-col chunks; +src residual
  const int mr = wave >> 1, ncb = (wave & 1) * 2;  // wave grid 4 x 2, 2 n-tiles
  for (int c = 0; c < 8; ++c) {
    int d0 = c * 64;
    __syncthreads();
#pragma unroll
    for (int s = 0; s < 48; ++s) {           // stage V^T chunk: vt[nd][kk]
      int e = s * 256 + tid;
      int kk = e >> 6, nd = e & 63;
      int gt = w * Wn - Wn + kk;
      unsigned short val = 0;
      if (gt >= 0 && gt < Nn) val = vb[(bOff + gt) * Dm + d0 + nd];
      u.vt[nd * 192 + kk] = val;
    }
    __syncthreads();
    floatx8 oa[2] = {z, z};
#pragma unroll
    for (int ks = 0; ks < 6; ++ks) {
      int kk0 = ks * 32;
      Frag pa;
      load_frag_a(pa, pLds + kk0, mr * 16, 192, lane);
#pragma unroll
      for (int tt = 0; tt < 2; ++tt) {
        Frag bfv;
        load_frag_b(bfv, u.vt + kk0, (ncb + tt) * 16, 192, lane);
        oa[tt] = wmma_bf16(pa, bfv, oa[tt]);
      }
    }
#pragma unroll
    for (int tt = 0; tt < 2; ++tt)
#pragma unroll
      for (int i = 0; i < 8; ++i) {
        int m = mr * 16 + h * 8 + i;
        int col = d0 + (ncb + tt) * 16 + nn;
        size_t idx = (bOff + (size_t)w * Wn + m) * Dm + col;
        outres[idx] = oa[tt][i] + src[idx];
      }
  }
}

// ---- layernorm over D=512, one block per row; optional residual add --------
__global__ __launch_bounds__(256) void ln_kernel(
    const float* __restrict__ in, const float* __restrict__ add,
    const float* __restrict__ g, const float* __restrict__ bta,
    float* __restrict__ out)
{
  const int row = blockIdx.x, tid = threadIdx.x;
  const size_t base = (size_t)row * 512;
  float x0 = in[base + tid];
  float x1 = in[base + tid + 256];
  if (add) { x0 += add[base + tid]; x1 += add[base + tid + 256]; }
  __shared__ float red[8];
  float s = x0 + x1;
#pragma unroll
  for (int off = 16; off >= 1; off >>= 1) s += __shfl_xor(s, off);
  if ((tid & 31) == 0) red[tid >> 5] = s;
  __syncthreads();
  float tot = 0.f;
#pragma unroll
  for (int i = 0; i < 8; ++i) tot += red[i];
  float mean = tot * (1.0f / 512.0f);
  __syncthreads();
  float d0 = x0 - mean, d1 = x1 - mean;
  float vs = d0 * d0 + d1 * d1;
#pragma unroll
  for (int off = 16; off >= 1; off >>= 1) vs += __shfl_xor(vs, off);
  if ((tid & 31) == 0) red[tid >> 5] = vs;
  __syncthreads();
  float vtot = 0.f;
#pragma unroll
  for (int i = 0; i < 8; ++i) vtot += red[i];
  float rs = rsqrtf(vtot * (1.0f / 512.0f) + 1e-5f);
  out[base + tid]       = d0 * rs * g[tid] + bta[tid];
  out[base + tid + 256] = d1 * rs * g[tid + 256] + bta[tid + 256];
}

// ---- launcher --------------------------------------------------------------
extern "C" void kernel_launch(void* const* d_in, const int* in_sizes, int n_in,
                              void* d_out, int out_size, void* d_ws, size_t ws_size,
                              hipStream_t stream) {
  (void)in_sizes; (void)n_in; (void)out_size; (void)ws_size;
  const float* src  = (const float*)d_in[0];
  const unsigned char* mask = (const unsigned char*)d_in[1];
  const float* Wq = (const float*)d_in[2];  const float* bq = (const float*)d_in[3];
  const float* Wk = (const float*)d_in[4];  const float* bk = (const float*)d_in[5];
  const float* Wv = (const float*)d_in[6];  const float* bv = (const float*)d_in[7];
  const float* ln1g = (const float*)d_in[8];  const float* ln1b = (const float*)d_in[9];
  const float* W1 = (const float*)d_in[10]; const float* b1 = (const float*)d_in[11];
  const float* W2 = (const float*)d_in[12]; const float* b2 = (const float*)d_in[13];
  const float* ln2g = (const float*)d_in[14]; const float* ln2b = (const float*)d_in[15];

  const size_t Mrows = 4ull * 8192;               // B*N = 32768 rows
  char* ws = (char*)d_ws;
  const size_t QKV = Mrows * 512 * sizeof(unsigned short);   // 32 MB each
  unsigned short* qbuf = (unsigned short*)ws;
  unsigned short* kbuf = (unsigned short*)(ws + QKV);
  unsigned short* vbuf = (unsigned short*)(ws + 2 * QKV);
  float* attnres = (float*)(ws + 3 * QKV);                          // 64 MB
  float* xbuf    = (float*)(ws + 3 * QKV + Mrows * 512 * sizeof(float)); // 64 MB
  unsigned short* hbuf = (unsigned short*)ws;   // 128 MB, overlays q/k/v/attnres
  float* ff2out  = (float*)(ws + 3 * QKV + 2 * Mrows * 512 * sizeof(float));

  dim3 blk(256);
  const float qscale = 0.04419417382415922f;      // 512^-0.5

  // QKV projections -> bf16 (q pre-scaled)
  gemm_kernel<0, false><<<dim3(256, 4), blk, 0, stream>>>(src, Wq, bq, qbuf, 512, 512, qscale);
  gemm_kernel<0, false><<<dim3(256, 4), blk, 0, stream>>>(src, Wk, bk, kbuf, 512, 512, 1.0f);
  gemm_kernel<0, false><<<dim3(256, 4), blk, 0, stream>>>(src, Wv, bv, vbuf, 512, 512, 1.0f);
  // local attention (+ src residual fused)
  attn_kernel<<<dim3(128, 4), blk, 0, stream>>>(qbuf, kbuf, vbuf, src, mask, attnres);
  // x = LN(src + attn)
  ln_kernel<<<dim3((unsigned)Mrows), blk, 0, stream>>>(attnres, nullptr, ln1g, ln1b, xbuf);
  // FF1: relu(x @ W1 + b1) -> bf16 hidden
  gemm_kernel<1, false><<<dim3(256, 16), blk, 0, stream>>>(xbuf, W1, b1, hbuf, 512, 2048, 1.0f);
  // FF2: hidden @ W2 + b2 -> f32
  gemm_kernel<2, true><<<dim3(256, 4), blk, 0, stream>>>(hbuf, W2, b2, ff2out, 2048, 512, 1.0f);
  // out = LN(x + ff)
  ln_kernel<<<dim3((unsigned)Mrows), blk, 0, stream>>>(xbuf, ff2out, ln2g, ln2b, (float*)d_out);
}
